// Head_48790828483176
// MI455X (gfx1250) — matrix operand order repaired and checked by reading, not compiled, not run on hardware
//
#include <hip/hip_runtime.h>

// CDNA5 / gfx1250 flash-attention for a single 128-dim head.
// B=8, T=2048, C=128. All matmuls on v_wmma_f32_16x16x32_bf16 (wave32).
// K-block staging uses GLOBAL_LOAD_ASYNC_TO_LDS_B128 (ASYNCcnt) when available.

typedef __attribute__((ext_vector_type(16))) __bf16 v16bf;
typedef __attribute__((ext_vector_type(8)))  __bf16 v8bf;
typedef __attribute__((ext_vector_type(8)))  float  v8f;

#define C_DIM 128
#define T_DIM 2048
#define B_DIM 8
#define QT    64      // query rows per workgroup (4 waves x 16)
#define KB    32      // key rows per block (two 16-wide N tiles)
#define VPAD  48      // transposed-V row length: 96 B, 32B-aligned rows

#ifndef __has_builtin
#define __has_builtin(x) 0
#endif
#if __has_builtin(__builtin_amdgcn_global_load_async_to_lds_b128)
#define USE_ASYNC_LDS 1
#else
#define USE_ASYNC_LDS 0
#endif

#if USE_ASYNC_LDS
// Signature (from clang diagnostic): (int4 AS1* src, int4 AS3* dst, imm, imm)
typedef int v4i_ __attribute__((vector_size(16)));
typedef __attribute__((address_space(1))) v4i_ as1_v4i;
typedef __attribute__((address_space(3))) v4i_ as3_v4i;
#endif

__device__ __forceinline__ v8f wmma_bf16(v16bf a, v16bf b, v8f c) {
  // (neg_a, A, neg_b, B, c_mod, C, reuse_a, reuse_b)
  return __builtin_amdgcn_wmma_f32_16x16x32_bf16(false, a, false, b,
                                                 (short)0, c, false, false);
}

// ---------------------------------------------------------------------------
// Projection: y = x @ W^T, 16x128 tile per wave.
// A (16x32 bf16): lane l -> row M=l%16; element e -> K = e + 16*(l/16).
// B (32x16 bf16): lane l -> col N=l%16; element e -> K = e + 16*(l/16).
//   B[c,d] = W[d,c]  =>  each lane reads a contiguous row of W (no transpose).
// C/D (16x16 f32): VGPR r, lane l -> (M = r + 8*(l/16), N = l%16).
// ---------------------------------------------------------------------------
__global__ __launch_bounds__(32)
void proj_qkv(const float* __restrict__ x,
              const float* __restrict__ Wk,
              const float* __restrict__ Wq,
              const float* __restrict__ Wv,
              __bf16* __restrict__ kws,
              __bf16* __restrict__ qws,
              __bf16* __restrict__ vws) {
  const int lane = threadIdx.x & 31;
  const int half = lane >> 4;
  const int lm   = lane & 15;
  const int t0   = blockIdx.x * 16;   // row tile over flattened [B*T]

  const float* W;
  __bf16* y;
  if (blockIdx.y == 0)      { W = Wk; y = kws; }
  else if (blockIdx.y == 1) { W = Wq; y = qws; }
  else                      { W = Wv; y = vws; }

  // x tile as 4 A-fragments (K = 128 in 4 chunks of 32); f32 loads + packed cvt.
  v16bf a[4];
  const float* xrow = x + (size_t)(t0 + lm) * C_DIM;
#pragma unroll
  for (int kc = 0; kc < 4; ++kc) {
    const int c0 = kc * 32 + half * 16;
#pragma unroll
    for (int e = 0; e < 8; ++e) {
      a[kc][e]     = (__bf16)xrow[c0 + e];
      a[kc][e + 8] = (__bf16)xrow[c0 + 8 + e];
    }
  }

#pragma unroll
  for (int dt = 0; dt < 8; ++dt) {      // 8 tiles of 16 output columns
    v8f acc = {};
    const float* wrow = W + (size_t)(dt * 16 + lm) * C_DIM + half * 16;
#pragma unroll
    for (int kc = 0; kc < 4; ++kc) {
      v16bf bf;
#pragma unroll
      for (int e = 0; e < 16; ++e) bf[e] = (__bf16)wrow[kc * 32 + e];
      acc = wmma_bf16(a[kc], bf, acc);
    }
#pragma unroll
    for (int r = 0; r < 8; ++r)
      y[(size_t)(t0 + r + 8 * half) * C_DIM + dt * 16 + lm] = (__bf16)acc[r];
  }
}

// ---------------------------------------------------------------------------
// Flash attention: 4 waves per workgroup; wave w owns 16 query rows.
// K/V blocks of 32 keys staged in LDS, shared by the 4 waves.
// ---------------------------------------------------------------------------
__global__ __launch_bounds__(128)
void attn_flash(const __bf16* __restrict__ qg,
                const __bf16* __restrict__ kg,
                const __bf16* __restrict__ vg,
                float* __restrict__ out) {
  __shared__ __align__(32) __bf16 ks[KB][C_DIM];       // K block, row-major
  __shared__ __align__(32) __bf16 vts[C_DIM][VPAD];    // V block, transposed [d][key]
  __shared__ __align__(32) __bf16 ps[4][16][KB];       // per-wave P scratch

  const int tid  = threadIdx.x;
  const int lane = tid & 31;
  const int wave = tid >> 5;
  const int half = lane >> 4;
  const int lm   = lane & 15;

  const int b     = blockIdx.y;
  const int qt0   = blockIdx.x * QT;
  const int qbase = qt0 + wave * 16;

  const __bf16* Q = qg + ((size_t)b * T_DIM + qbase) * C_DIM;
  const __bf16* K = kg + (size_t)b * T_DIM * C_DIM;
  const __bf16* V = vg + (size_t)b * T_DIM * C_DIM;

  // Q tile as 4 A-fragments, loaded as two aligned 16B vectors each.
  v16bf qa[4];
  {
    const __bf16* qrow = Q + (size_t)lm * C_DIM;
#pragma unroll
    for (int kc = 0; kc < 4; ++kc) {
      const int c0 = kc * 32 + half * 16;
      v8bf lo = *(const v8bf*)(qrow + c0);
      v8bf hi = *(const v8bf*)(qrow + c0 + 8);
      qa[kc] = __builtin_shufflevector(lo, hi, 0, 1, 2, 3, 4, 5, 6, 7,
                                       8, 9, 10, 11, 12, 13, 14, 15);
    }
  }

  v8f o[8];
#pragma unroll
  for (int dt = 0; dt < 8; ++dt) { v8f z = {}; o[dt] = z; }

  // Online-softmax stats: element r belongs to row (r + 8*half), like C-layout.
  float m[8], lsum[8];
#pragma unroll
  for (int r = 0; r < 8; ++r) { m[r] = -1e30f; lsum[r] = 0.0f; }

  const float scale = 0.08838834764831845f;   // 128^-0.5
  const int nblocks = qt0 / KB + 2;           // causal coverage for rows qt0..qt0+63

  for (int sb = 0; sb < nblocks; ++sb) {
    const int s0 = sb * KB;

    __syncthreads();   // previous iteration's LDS readers are done

    // Cooperative staging: K rows async->LDS; V loaded and transposed into LDS.
    for (int i = tid; i < KB * 8; i += 128) {   // 8 x 16-element (32B) segments/row
      const int row = i >> 3, seg = i & 7;
      const __bf16* gk = K + (size_t)(s0 + row) * C_DIM + seg * 16;
      __bf16*       lk = &ks[row][seg * 16];
#if USE_ASYNC_LDS
      __builtin_amdgcn_global_load_async_to_lds_b128(
          (as1_v4i*)(uintptr_t)gk, (as3_v4i*)(uint32_t)(uintptr_t)lk, 0, 0);
      __builtin_amdgcn_global_load_async_to_lds_b128(
          (as1_v4i*)(uintptr_t)(gk + 8), (as3_v4i*)(uint32_t)(uintptr_t)(lk + 8), 0, 0);
#else
      *(uint4*)lk = *(const uint4*)gk;
#endif
      const __bf16* vsrc = V + (size_t)(s0 + row) * C_DIM + seg * 16;
#pragma unroll
      for (int e = 0; e < 16; ++e) vts[seg * 16 + e][row] = vsrc[e];
    }
#if USE_ASYNC_LDS
#if __has_builtin(__builtin_amdgcn_s_wait_asynccnt)
    __builtin_amdgcn_s_wait_asynccnt(0);
#else
    asm volatile("s_wait_asynccnt 0" ::: "memory");
#endif
#endif
    __syncthreads();

    if (s0 <= qbase + 15) {   // wave-uniform: skip fully masked key blocks
      // ---- S = Q K^T for two 16-wide key tiles ----
      v8f s[2];
#pragma unroll
      for (int nt = 0; nt < 2; ++nt) {
        v8f acc = {};
#pragma unroll
        for (int kc = 0; kc < 4; ++kc) {
          v16bf bf = *(const v16bf*)&ks[nt * 16 + lm][kc * 32 + 16 * half];
          acc = wmma_bf16(qa[kc], bf, acc);
        }
        s[nt] = acc;
      }

      // ---- scale + causal mask ----
#pragma unroll
      for (int nt = 0; nt < 2; ++nt)
#pragma unroll
        for (int r = 0; r < 8; ++r) {
          const int row = qbase + r + 8 * half;
          const int col = s0 + nt * 16 + lm;
          s[nt][r] = (col <= row) ? s[nt][r] * scale : -1e30f;
        }

      // ---- row max across the 16 lanes of each half-wave ----
      float rm[8];
#pragma unroll
      for (int r = 0; r < 8; ++r) {
        float t = fmaxf(s[0][r], s[1][r]);
#pragma unroll
        for (int off = 1; off < 16; off <<= 1)
          t = fmaxf(t, __shfl_xor(t, off, 32));
        rm[r] = t;
      }

      // ---- rescale running state ----
      float fac[8];
#pragma unroll
      for (int r = 0; r < 8; ++r) {
        const float mn = fmaxf(m[r], rm[r]);
        fac[r] = __expf(m[r] - mn);
        m[r] = mn;
      }
#pragma unroll
      for (int dt = 0; dt < 8; ++dt)
#pragma unroll
        for (int r = 0; r < 8; ++r) o[dt][r] *= fac[r];
#pragma unroll
      for (int r = 0; r < 8; ++r) lsum[r] *= fac[r];

      // ---- P = exp(S - m); accumulate row sums ----
#pragma unroll
      for (int nt = 0; nt < 2; ++nt)
#pragma unroll
        for (int r = 0; r < 8; ++r) s[nt][r] = __expf(s[nt][r] - m[r]);
#pragma unroll
      for (int r = 0; r < 8; ++r) {
        float t = s[0][r] + s[1][r];
#pragma unroll
        for (int off = 1; off < 16; off <<= 1)
          t += __shfl_xor(t, off, 32);
        lsum[r] += t;
      }

      // ---- C-layout f32 -> A-layout bf16 via per-wave LDS scratch ----
      // LDS ops of one wave complete in order (DScnt); fence the compiler only.
#pragma unroll
      for (int nt = 0; nt < 2; ++nt)
#pragma unroll
        for (int r = 0; r < 8; ++r)
          ps[wave][r + 8 * half][nt * 16 + lm] = (__bf16)s[nt][r];
      asm volatile("" ::: "memory");
      v16bf pa;
      {
        const __bf16* prow = &ps[wave][lm][half * 16];
        v8bf lo = *(const v8bf*)prow;
        v8bf hi = *(const v8bf*)(prow + 8);
        pa = __builtin_shufflevector(lo, hi, 0, 1, 2, 3, 4, 5, 6, 7,
                                     8, 9, 10, 11, 12, 13, 14, 15);
      }

      // ---- O += P @ V (transposed V: lane-contiguous B operand) ----
#pragma unroll
      for (int dt = 0; dt < 8; ++dt) {
        v16bf bf = *(const v16bf*)&vts[dt * 16 + lm][16 * half];
        o[dt] = wmma_bf16(pa, bf, o[dt]);
      }
    }
  }

  // ---- epilogue: normalize and store f32 output ----
  float inv[8];
#pragma unroll
  for (int r = 0; r < 8; ++r) inv[r] = 1.0f / lsum[r];   // >= 1 (diagonal present)

  float* orow = out + ((size_t)b * T_DIM + qbase) * C_DIM;
#pragma unroll
  for (int dt = 0; dt < 8; ++dt)
#pragma unroll
    for (int r = 0; r < 8; ++r)
      orow[(size_t)(r + 8 * half) * C_DIM + dt * 16 + lm] = o[dt][r] * inv[r];
}

// ---------------------------------------------------------------------------
extern "C" void kernel_launch(void* const* d_in, const int* in_sizes, int n_in,
                              void* d_out, int out_size, void* d_ws, size_t ws_size,
                              hipStream_t stream) {
  const float* x  = (const float*)d_in[0];
  const float* Wk = (const float*)d_in[1];
  const float* Wq = (const float*)d_in[2];
  const float* Wv = (const float*)d_in[3];
  float* out = (float*)d_out;

  // Workspace: bf16 k/q/v, each B*T*C = 2M elements = 4 MB -> 12 MB total.
  const size_t elems = (size_t)B_DIM * T_DIM * C_DIM;
  __bf16* kws = (__bf16*)d_ws;
  __bf16* qws = kws + elems;
  __bf16* vws = qws + elems;

  proj_qkv<<<dim3((B_DIM * T_DIM) / 16, 3), 32, 0, stream>>>(
      x, Wk, Wq, Wv, kws, qws, vws);

  attn_flash<<<dim3(T_DIM / QT, B_DIM), 128, 0, stream>>>(qws, kws, vws, out);
}
